// ConvGramClassifier_26774826123691
// MI455X (gfx1250) — matrix operand order repaired and checked by reading, not compiled, not run on hardware
//
#include <hip/hip_runtime.h>
#include <math.h>

#define TT    2048
#define EE    64
#define FF    128
#define CHUNK 256
#define ROWS  (CHUNK + 2)
#define LSTR  72            // padded LDS row stride (bf16 elements) -> 144B, conflict-free

typedef __attribute__((ext_vector_type(16))) __bf16       v16bf;
typedef __attribute__((ext_vector_type(8)))  float        v8f;
typedef __attribute__((ext_vector_type(4)))  unsigned int v4u;
typedef __attribute__((ext_vector_type(4)))  int          v4i;

#if __has_builtin(__builtin_amdgcn_global_load_async_to_lds_b128)
#define USE_ASYNC_LDS 1
#else
#define USE_ASYNC_LDS 0
#endif

union Frag { v16bf v; unsigned short s[16]; v4u q[2]; };

__device__ __forceinline__ unsigned short f2bf(float x) {
  unsigned int u = __float_as_uint(x);
  u += 0x7FFFu + ((u >> 16) & 1u);          // round-to-nearest-even
  return (unsigned short)(u >> 16);
}

__global__ void zero_kernel(float* p, int n) {
  int i = blockIdx.x * blockDim.x + threadIdx.x;
  if (i < n) p[i] = 0.0f;
}

// One-time fp32 -> bf16 conversion of the embedding table (6.4MB, L2-resident).
__global__ void cvt_table_kernel(const float* __restrict__ src,
                                 unsigned short* __restrict__ dst, int n) {
  int j = (blockIdx.x * blockDim.x + threadIdx.x) * 2;
  if (j < n) {     // n is even
    unsigned int lo = f2bf(src[j]);
    unsigned int hi = f2bf(src[j + 1]);
    *(unsigned int*)(dst + j) = lo | (hi << 16);
  }
}

__global__ __launch_bounds__(256)
void conv_ngram_kernel(const int*   __restrict__ tokens,
                       const int*   __restrict__ lengths,
                       const float* __restrict__ etab,
                       const unsigned short* __restrict__ btab,
                       const float* __restrict__ w1, const float* __restrict__ b1,
                       const float* __restrict__ w2, const float* __restrict__ b2,
                       const float* __restrict__ w3, const float* __restrict__ b3,
                       float*       __restrict__ feats)
{
  __shared__ unsigned short lds[ROWS * LSTR];

  const int b    = blockIdx.y;
  const int c0   = blockIdx.x * CHUNK;
  const int tid  = threadIdx.x;
  const int lane = tid & 31;
  const int wave = tid >> 5;
  const int g    = lane >> 4;     // lane half (0/1)
  const int fl   = lane & 15;
  const int f    = wave * 16 + fl;   // filter owned by this lane pair
  const int len  = lengths[b];

#if USE_ASYNC_LDS
  (void)etab;
  // ---- Async global->LDS gather of bf16 rows: 8 x 16B per row, no VGPR transit.
  for (int base = 0; base < ROWS; base += 32) {
    int r = base + (tid >> 3);
    if (r < ROWS) {
      int row = c0 + r; if (row > TT - 1) row = TT - 1;   // clamp; tail is masked out
      int tok = tokens[b * TT + row];
      int ch  = tid & 7;
      __builtin_amdgcn_global_load_async_to_lds_b128(
          (v4i*)(btab + (size_t)tok * EE + ch * 8),
          (v4i*)&lds[r * LSTR + ch * 8], 0, 0);
    }
  }
#else
  (void)btab;
  // ---- Fallback: fp32 gather + inline convert (4 threads/row, 16 elems each)
  for (int base = 0; base < ROWS; base += 64) {
    int r = base + (tid >> 2);
    if (r < ROWS) {
      int row = c0 + r; if (row > TT - 1) row = TT - 1;
      int tok = tokens[b * TT + row];
      const float4* s4 = (const float4*)(etab + (long)tok * EE + (tid & 3) * 16);
      unsigned int* d32 = (unsigned int*)&lds[r * LSTR + (tid & 3) * 16];
      #pragma unroll
      for (int j = 0; j < 4; j++) {
        float4 vv = s4[j];
        d32[2 * j + 0] = (unsigned int)f2bf(vv.x) | ((unsigned int)f2bf(vv.y) << 16);
        d32[2 * j + 1] = (unsigned int)f2bf(vv.z) | ((unsigned int)f2bf(vv.w) << 16);
      }
    }
  }
#endif

  // ---- B fragments (weights, bf16). 16-bit B KxN layout: lane -> column N=fl,
  //      K = kh*32 + 8*g + i (i<8), kh*32 + 16 + 8*g + (i-8) (i>=8)
  auto loadB = [&](const float* w, int tap, int kh) -> v16bf {
    Frag fr;
    int kb = kh * 32 + g * 8;
    #pragma unroll
    for (int i = 0; i < 16; i++)
      fr.s[i] = f2bf(w[(tap * EE + kb + (i < 8 ? i : i + 8)) * FF + f]);
    return fr.v;
  };
  v16bf B1[2], B2[2][2], B3[3][2];
  #pragma unroll
  for (int kh = 0; kh < 2; kh++) {
    B1[kh]    = loadB(w1, 0, kh);
    B2[0][kh] = loadB(w2, 0, kh);
    B2[1][kh] = loadB(w2, 1, kh);
    B3[0][kh] = loadB(w3, 0, kh);
    B3[1][kh] = loadB(w3, 1, kh);
    B3[2][kh] = loadB(w3, 2, kh);
  }
  const float bias1 = b1[f], bias2 = b2[f], bias3 = b3[f];

#if USE_ASYNC_LDS
#if __has_builtin(__builtin_amdgcn_s_wait_asynccnt)
  __builtin_amdgcn_s_wait_asynccnt(0);
#else
  asm volatile("s_wait_asynccnt 0" ::: "memory");
#endif
#endif
  __syncthreads();

  // ---- A fragment (16-bit A 16x32 layout): lane row M=fl,
  //      V0-3: K = kOff+8g+0..7 ; V4-7: K = kOff+16+8g+0..7  -> two 16B ds loads
  auto loadA = [&](int rowBase, int kh) -> v16bf {
    Frag fr;
    const unsigned short* p = &lds[(rowBase + fl) * LSTR + kh * 32 + g * 8];
    fr.q[0] = *(const v4u*)(p);
    fr.q[1] = *(const v4u*)(p + 16);
    return fr.v;
  };

  // bias+relu -> ragged mask (zero) -> maxpool(4) with VALID window drop -> sum
  auto pool = [&](const v8f& c, float bias, int tBase, int n) -> float {
    float acc = 0.0f, m0 = 0.0f, m1 = 0.0f;
    #pragma unroll
    for (int v = 0; v < 4; v++) {
      float val = fmaxf(c[v] + bias, 0.0f);
      if (tBase + v >= len - n + 1) val = 0.0f;   // ragged mask (included in max as 0)
      m0 = fmaxf(m0, val);
    }
    #pragma unroll
    for (int v = 4; v < 8; v++) {
      float val = fmaxf(c[v] + bias, 0.0f);
      if (tBase + v >= len - n + 1) val = 0.0f;
      m1 = fmaxf(m1, val);
    }
    if (tBase + 3 < TT - n + 1) acc += m0;        // VALID pooling: drop partial windows
    if (tBase + 7 < TT - n + 1) acc += m1;
    return acc;
  };

  float sum1 = 0.f, sum2 = 0.f, sum3 = 0.f;
  for (int m = 0; m < CHUNK / 16; m++) {
    int rb = m * 16;
    v16bf A00 = loadA(rb,     0), A01 = loadA(rb,     1);
    v16bf A10 = loadA(rb + 1, 0), A11 = loadA(rb + 1, 1);
    v16bf A20 = loadA(rb + 2, 0), A21 = loadA(rb + 2, 1);

    v8f c1 = {0.f,0.f,0.f,0.f,0.f,0.f,0.f,0.f};
    v8f c2 = c1, c3 = c1;

    c1 = __builtin_amdgcn_wmma_f32_16x16x32_bf16(false, A00, false, B1[0],    (short)0, c1, false, false);
    c1 = __builtin_amdgcn_wmma_f32_16x16x32_bf16(false, A01, false, B1[1],    (short)0, c1, false, false);

    c2 = __builtin_amdgcn_wmma_f32_16x16x32_bf16(false, A00, false, B2[0][0], (short)0, c2, false, false);
    c2 = __builtin_amdgcn_wmma_f32_16x16x32_bf16(false, A01, false, B2[0][1], (short)0, c2, false, false);
    c2 = __builtin_amdgcn_wmma_f32_16x16x32_bf16(false, A10, false, B2[1][0], (short)0, c2, false, false);
    c2 = __builtin_amdgcn_wmma_f32_16x16x32_bf16(false, A11, false, B2[1][1], (short)0, c2, false, false);

    c3 = __builtin_amdgcn_wmma_f32_16x16x32_bf16(false, A00, false, B3[0][0], (short)0, c3, false, false);
    c3 = __builtin_amdgcn_wmma_f32_16x16x32_bf16(false, A01, false, B3[0][1], (short)0, c3, false, false);
    c3 = __builtin_amdgcn_wmma_f32_16x16x32_bf16(false, A10, false, B3[1][0], (short)0, c3, false, false);
    c3 = __builtin_amdgcn_wmma_f32_16x16x32_bf16(false, A11, false, B3[1][1], (short)0, c3, false, false);
    c3 = __builtin_amdgcn_wmma_f32_16x16x32_bf16(false, A20, false, B3[2][0], (short)0, c3, false, false);
    c3 = __builtin_amdgcn_wmma_f32_16x16x32_bf16(false, A21, false, B3[2][1], (short)0, c3, false, false);

    // C/D layout: lane holds rows M = 8*g + v (v = VGPR index), column N = fl
    int tBase = c0 + rb + 8 * g;
    sum1 += pool(c1, bias1, tBase, 1);
    sum2 += pool(c2, bias2, tBase, 2);
    sum3 += pool(c3, bias3, tBase, 3);
  }

  atomicAdd(&feats[b * 384 + 0 * FF + f], sum1);
  atomicAdd(&feats[b * 384 + 1 * FF + f], sum2);
  atomicAdd(&feats[b * 384 + 2 * FF + f], sum3);
}

__global__ __launch_bounds__(128)
void finalize_kernel(const float* __restrict__ feats,
                     const float* __restrict__ svm_w, const float* __restrict__ svm_b,
                     const float* __restrict__ pa,    const float* __restrict__ pb,
                     float* __restrict__ out)
{
  __shared__ float x[384];
  __shared__ float red[128];
  const int b = blockIdx.x, tid = threadIdx.x;

  float ss = 0.0f;
  for (int i = tid; i < 384; i += 128) {
    float v = feats[b * 384 + i];
    x[i] = v;
    ss += v * v;
  }
  red[tid] = ss;
  __syncthreads();
  for (int s = 64; s > 0; s >>= 1) {
    if (tid < s) red[tid] += red[tid + s];
    __syncthreads();
  }
  float inv = rsqrtf(fmaxf(red[0], 1e-12f));

  if (tid < 10) {
    float s = svm_b[tid];
    for (int i = 0; i < 384; i++) s += x[i] * inv * svm_w[i * 10 + tid];
    float z = s * pa[tid] + pb[tid];
    out[b * 10 + tid] = 1.0f / (1.0f + expf(-z));
  }
}

extern "C" void kernel_launch(void* const* d_in, const int* in_sizes, int n_in,
                              void* d_out, int out_size, void* d_ws, size_t ws_size,
                              hipStream_t stream) {
  const int*   tokens  = (const int*)d_in[0];
  const int*   lengths = (const int*)d_in[1];
  const float* etab    = (const float*)d_in[2];
  const float* w1 = (const float*)d_in[3];
  const float* b1 = (const float*)d_in[4];
  const float* w2 = (const float*)d_in[5];
  const float* b2 = (const float*)d_in[6];
  const float* w3 = (const float*)d_in[7];
  const float* b3 = (const float*)d_in[8];
  const float* svm_w = (const float*)d_in[9];
  const float* svm_b = (const float*)d_in[10];
  const float* pa = (const float*)d_in[11];
  const float* pb = (const float*)d_in[12];
  float* out   = (float*)d_out;

  float*          feats = (float*)d_ws;                       // 64*384 f32
  unsigned short* btab  = (unsigned short*)((char*)d_ws + (128 << 10)); // bf16 table

  const int nfeat = 64 * 384;
  const int ntab  = in_sizes[2];                              // 50000*64
  zero_kernel<<<(nfeat + 255) / 256, 256, 0, stream>>>(feats, nfeat);
  cvt_table_kernel<<<(ntab / 2 + 255) / 256, 256, 0, stream>>>(etab, btab, ntab);
  conv_ngram_kernel<<<dim3(TT / CHUNK, 64), 256, 0, stream>>>(
      tokens, lengths, etab, btab, w1, b1, w2, b2, w3, b3, feats);
  finalize_kernel<<<64, 128, 0, stream>>>(feats, svm_w, svm_b, pa, pb, out);
}
